// MSAColumnAttention_26242250179308
// MI455X (gfx1250) — matrix operand order repaired and checked by reading, not compiled, not run on hardware
//
#include <hip/hip_runtime.h>

// ---------- types & helpers ----------
typedef __attribute__((ext_vector_type(16))) __bf16 v16bf;
typedef __attribute__((ext_vector_type(8)))  float  v8f;
typedef __attribute__((ext_vector_type(4)))  int    v4i_;

union Frag32 { uint4 q[2]; v16bf v; };   // 32B = 16 bf16 per lane (A or B fragment)

static __device__ __forceinline__ unsigned short f2bf_bits(float f) {
  unsigned int u = __float_as_uint(f);
  u += 0x7FFFu + ((u >> 16) & 1u);       // round-to-nearest-even
  return (unsigned short)(u >> 16);
}
static __device__ __forceinline__ float bf2f(unsigned short b) {
  return __uint_as_float(((unsigned int)b) << 16);
}
static __device__ __forceinline__ unsigned int pack2(float a, float b) {
  return (unsigned int)f2bf_bits(a) | ((unsigned int)f2bf_bits(b) << 16);
}

// ---- gfx1250 async global->LDS copy (ASYNCcnt path), with safe fallback ----
#if defined(__AMDGCN__) && \
    __has_builtin(__builtin_amdgcn_global_load_async_to_lds_b128) && \
    __has_builtin(__builtin_amdgcn_s_wait_asynccnt)
#define USE_ASYNC_LDS 1
#else
#define USE_ASYNC_LDS 0
#endif

// copy one 64-byte row (32 bf16) from global to LDS
static __device__ __forceinline__ void stage_row64(unsigned short* dst,
                                                   const unsigned short* src) {
#if USE_ASYNC_LDS
  __attribute__((address_space(1))) v4i_* g =
      (__attribute__((address_space(1))) v4i_*)src;
  __attribute__((address_space(3))) v4i_* l =
      (__attribute__((address_space(3))) v4i_*)dst;
  __builtin_amdgcn_global_load_async_to_lds_b128(g + 0, l + 0, 0, 0);
  __builtin_amdgcn_global_load_async_to_lds_b128(g + 1, l + 1, 0, 0);
  __builtin_amdgcn_global_load_async_to_lds_b128(g + 2, l + 2, 0, 0);
  __builtin_amdgcn_global_load_async_to_lds_b128(g + 3, l + 3, 0, 0);
#else
  *(uint4*)(dst + 0)  = *(const uint4*)(src + 0);
  *(uint4*)(dst + 8)  = *(const uint4*)(src + 8);
  *(uint4*)(dst + 16) = *(const uint4*)(src + 16);
  *(uint4*)(dst + 24) = *(const uint4*)(src + 24);
#endif
}
static __device__ __forceinline__ void stage_fence() {
#if USE_ASYNC_LDS
  __builtin_amdgcn_s_wait_asynccnt(0);
#endif
}

#define NSEQ 256
#define NRES 256
#define CDIM 256
#define HN   8
#define DH   32

// ---------- K0: weight prep (transpose + bf16) ----------
// Wt  [1024][256] : row n holds column n of [Wq|Wk|Wv|Wg] (K-contiguous)
// Wtf [ 256][256] : Wf with N-major layout
__global__ __launch_bounds__(256) void prep_weights(
    const float* __restrict__ Wq, const float* __restrict__ Wk,
    const float* __restrict__ Wv, const float* __restrict__ Wg,
    const float* __restrict__ Wf,
    unsigned short* __restrict__ Wt, unsigned short* __restrict__ Wtf)
{
  const int n = blockIdx.x;       // 0..1279
  const int k = threadIdx.x;      // 0..255
  if (n < 1024) {
    const float* W = (n < 256) ? Wq : (n < 512) ? Wk : (n < 768) ? Wv : Wg;
    const int col = n & 255;
    Wt[n * 256 + k] = f2bf_bits(W[k * 256 + col]);
  } else {
    const int col = n - 1024;
    Wtf[col * 256 + k] = f2bf_bits(Wf[k * 256 + col]);
  }
}

// ---------- K1: LayerNorm (wave per row) -> bf16 xn [65536][256] ----------
__global__ __launch_bounds__(256) void ln_kernel(
    const float* __restrict__ x, const float* __restrict__ gamma,
    const float* __restrict__ beta, unsigned short* __restrict__ xn)
{
  const int lane = threadIdx.x & 31;
  const int wave = threadIdx.x >> 5;
  const int row  = blockIdx.x * 8 + wave;          // 8192 blocks * 8 waves = 65536 rows
  const float* xr = x + row * 256 + lane * 8;
  float4 a = *(const float4*)(xr);
  float4 b = *(const float4*)(xr + 4);
  float s  = a.x + a.y + a.z + a.w + b.x + b.y + b.z + b.w;
  float sq = a.x*a.x + a.y*a.y + a.z*a.z + a.w*a.w
           + b.x*b.x + b.y*b.y + b.z*b.z + b.w*b.w;
  #pragma unroll
  for (int off = 16; off > 0; off >>= 1) {
    s  += __shfl_xor(s,  off, 32);
    sq += __shfl_xor(sq, off, 32);
  }
  const float mean = s  * (1.0f / 256.0f);
  const float var  = sq * (1.0f / 256.0f) - mean * mean;
  const float rstd = rsqrtf(var + 1e-5f);
  float4 g0 = *(const float4*)(gamma + lane * 8);
  float4 g1 = *(const float4*)(gamma + lane * 8 + 4);
  float4 b0 = *(const float4*)(beta  + lane * 8);
  float4 b1 = *(const float4*)(beta  + lane * 8 + 4);
  float o0 = (a.x - mean) * rstd * g0.x + b0.x;
  float o1 = (a.y - mean) * rstd * g0.y + b0.y;
  float o2 = (a.z - mean) * rstd * g0.z + b0.z;
  float o3 = (a.w - mean) * rstd * g0.w + b0.w;
  float o4 = (b.x - mean) * rstd * g1.x + b1.x;
  float o5 = (b.y - mean) * rstd * g1.y + b1.y;
  float o6 = (b.z - mean) * rstd * g1.z + b1.z;
  float o7 = (b.w - mean) * rstd * g1.w + b1.w;
  uint4 o;
  o.x = pack2(o0, o1); o.y = pack2(o2, o3);
  o.z = pack2(o4, o5); o.w = pack2(o6, o7);
  *(uint4*)(xn + row * 256 + lane * 8) = o;
}

// ---------- K2: fused QKVG GEMM: [65536,256] x [256,1024] (bf16 WMMA) ----------
__global__ __launch_bounds__(256) void gemm_qkvg(
    const unsigned short* __restrict__ xn,
    const unsigned short* __restrict__ Wt,
    const float* __restrict__ bg,
    unsigned short* __restrict__ qkvg)
{
  const int lane  = threadIdx.x & 31;
  const int wave  = threadIdx.x >> 5;
  const int half  = lane >> 4;
  const int l16   = lane & 15;
  const int Mbase = blockIdx.x * 128 + wave * 16;
  const int Nbase = blockIdx.y * 64;

  v8f acc[4] = {};
  const unsigned short* aptr = xn + (Mbase + l16) * 256 + half * 8;
  const unsigned short* bptr = Wt + (Nbase + l16) * 256 + half * 8;
  __builtin_prefetch(aptr, 0, 1);   // -> global_prefetch_b8
  __builtin_prefetch(bptr, 0, 1);

  #pragma unroll
  for (int ks = 0; ks < 8; ++ks) {
    Frag32 a;
    a.q[0] = *(const uint4*)(aptr + ks * 32);
    a.q[1] = *(const uint4*)(aptr + ks * 32 + 16);
    Frag32 b[4];
    #pragma unroll
    for (int j = 0; j < 4; ++j) {
      const unsigned short* bp = bptr + j * 4096 + ks * 32;
      b[j].q[0] = *(const uint4*)(bp);
      b[j].q[1] = *(const uint4*)(bp + 16);
    }
    #pragma unroll
    for (int j = 0; j < 4; ++j)
      acc[j] = __builtin_amdgcn_wmma_f32_16x16x32_bf16(
          false, a.v, false, b[j].v, (short)0, acc[j], false, false);
  }

  #pragma unroll
  for (int j = 0; j < 4; ++j) {
    const int n = Nbase + j * 16 + l16;
    const bool isGate = (n >= 768);
    const float bias = isGate ? bg[n - 768] : 0.0f;
    #pragma unroll
    for (int r = 0; r < 8; ++r) {
      const int row = Mbase + r + half * 8;
      float v = acc[j][r] + bias;
      if (isGate) v = 1.0f / (1.0f + __expf(-v));
      qkvg[row * 1024 + n] = f2bf_bits(v);
    }
  }
}

// ---------- K3: attention per (m, h, 64-query block) ----------
// qkvg layout: [i*256+m][ q(0..255) | k(256..511) | v(512..767) | g(768..1023) ]
__global__ __launch_bounds__(128) void attn_kernel(
    const unsigned short* __restrict__ qkvg,
    const float* __restrict__ mask,
    unsigned short* __restrict__ attn)
{
  __shared__ __attribute__((aligned(16))) unsigned short ks_[256][32];  // 16KB
  __shared__ __attribute__((aligned(16))) unsigned short vts[32][256];  // 16KB (v transposed)
  __shared__ __attribute__((aligned(16))) unsigned short qs [64][32];   // 4KB
  __shared__ __attribute__((aligned(16))) unsigned short gs [64][32];   // 4KB
  __shared__ __attribute__((aligned(16))) unsigned short wts[64][256];  // 32KB softmax weights
  __shared__ float bias_s[256];                                          // 1KB

  const int m    = blockIdx.x;
  const int h    = blockIdx.y;
  const int qb   = blockIdx.z;          // 64-row query block
  const int tid  = threadIdx.x;         // 0..127
  const int lane = tid & 31;
  const int wave = tid >> 5;            // 0..3
  const int half = lane >> 4;
  const int l16  = lane & 15;

  // stage K (async), V^T (register transpose), Q/G (async), mask bias
  for (int j = tid; j < 256; j += 128) {
    stage_row64(&ks_[j][0], qkvg + (j * 256 + m) * 1024 + 256 + h * 32);
    const unsigned short* vsrc = qkvg + (j * 256 + m) * 1024 + 512 + h * 32;
    uint4 w0 = *(const uint4*)(vsrc);
    uint4 w1 = *(const uint4*)(vsrc + 8);
    uint4 w2 = *(const uint4*)(vsrc + 16);
    uint4 w3 = *(const uint4*)(vsrc + 24);
    unsigned int wv[16] = {w0.x, w0.y, w0.z, w0.w, w1.x, w1.y, w1.z, w1.w,
                           w2.x, w2.y, w2.z, w2.w, w3.x, w3.y, w3.z, w3.w};
    #pragma unroll
    for (int d2 = 0; d2 < 16; ++d2) {
      vts[d2 * 2 + 0][j] = (unsigned short)(wv[d2] & 0xFFFFu);
      vts[d2 * 2 + 1][j] = (unsigned short)(wv[d2] >> 16);
    }
    bias_s[j] = 1e9f * (mask[j * 256 + m] - 1.0f);
  }
  for (int r = tid; r < 64; r += 128) {
    const int i = qb * 64 + r;
    stage_row64(&qs[r][0], qkvg + (i * 256 + m) * 1024 + h * 32);
    stage_row64(&gs[r][0], qkvg + (i * 256 + m) * 1024 + 768 + h * 32);
  }
  stage_fence();        // s_wait_asynccnt 0 (async engine done for this wave)
  __syncthreads();

  // Phase A: aff = (q/sqrt(DH)) . k^T   -- K = DH = 32: one WMMA per 16x16 tile
  v8f aff[16];
  {
    Frag32 a;
    const unsigned short* ap = &qs[wave * 16 + l16][half * 8];
    a.q[0] = *(const uint4*)(ap);
    a.q[1] = *(const uint4*)(ap + 16);
    #pragma unroll
    for (int t = 0; t < 16; ++t) {
      Frag32 b;
      const unsigned short* bp = &ks_[t * 16 + l16][half * 8];
      b.q[0] = *(const uint4*)(bp);
      b.q[1] = *(const uint4*)(bp + 16);
      v8f z = {};
      aff[t] = __builtin_amdgcn_wmma_f32_16x16x32_bf16(
          false, a.v, false, b.v, (short)0, z, false, false);
    }
  }

  // softmax over j (16 tiles x 16-lane halves); row = wave*16 + r + half*8
  const float scale = 0.17677669529663687f;  // 1/sqrt(32)
  #pragma unroll
  for (int r = 0; r < 8; ++r) {
    float mx = -1e30f;
    #pragma unroll
    for (int t = 0; t < 16; ++t) {
      float v = aff[t][r] * scale + bias_s[t * 16 + l16];
      aff[t][r] = v;
      mx = fmaxf(mx, v);
    }
    mx = fmaxf(mx, __shfl_xor(mx, 1, 32));
    mx = fmaxf(mx, __shfl_xor(mx, 2, 32));
    mx = fmaxf(mx, __shfl_xor(mx, 4, 32));
    mx = fmaxf(mx, __shfl_xor(mx, 8, 32));
    float s = 0.0f;
    #pragma unroll
    for (int t = 0; t < 16; ++t) {
      float e = __expf(aff[t][r] - mx);
      aff[t][r] = e;
      s += e;
    }
    s += __shfl_xor(s, 1, 32);
    s += __shfl_xor(s, 2, 32);
    s += __shfl_xor(s, 4, 32);
    s += __shfl_xor(s, 8, 32);
    const float inv = 1.0f / s;
    const int row = wave * 16 + r + half * 8;
    #pragma unroll
    for (int t = 0; t < 16; ++t)
      wts[row][t * 16 + l16] = f2bf_bits(aff[t][r] * inv);
  }
  __syncthreads();

  // Phase B: out = weights . v   (K = 256 over keys, N = 32 = DH)
  v8f o[2] = {};
  #pragma unroll
  for (int ksx = 0; ksx < 8; ++ksx) {
    Frag32 a;
    const unsigned short* ap = &wts[wave * 16 + l16][ksx * 32 + half * 8];
    a.q[0] = *(const uint4*)(ap);
    a.q[1] = *(const uint4*)(ap + 16);
    #pragma unroll
    for (int t = 0; t < 2; ++t) {
      Frag32 b;
      const unsigned short* bp = &vts[t * 16 + l16][ksx * 32 + half * 8];
      b.q[0] = *(const uint4*)(bp);
      b.q[1] = *(const uint4*)(bp + 16);
      o[t] = __builtin_amdgcn_wmma_f32_16x16x32_bf16(
          false, a.v, false, b.v, (short)0, o[t], false, false);
    }
  }

  // gate + store bf16 to attn [i*256+m][h*32+d]
  #pragma unroll
  for (int t = 0; t < 2; ++t) {
    const int d = t * 16 + l16;
    #pragma unroll
    for (int r = 0; r < 8; ++r) {
      const int rl = wave * 16 + r + half * 8;
      const int i  = qb * 64 + rl;
      const float g = bf2f(gs[rl][d]);
      attn[(i * 256 + m) * 256 + h * 32 + d] = f2bf_bits(o[t][r] * g);
    }
  }
}

// ---------- K4: output GEMM: [65536,256] x [256,256] + bf -> f32 out ----------
__global__ __launch_bounds__(256) void gemm_out(
    const unsigned short* __restrict__ attn,
    const unsigned short* __restrict__ Wtf,
    const float* __restrict__ bf,
    float* __restrict__ out)
{
  const int lane  = threadIdx.x & 31;
  const int wave  = threadIdx.x >> 5;
  const int half  = lane >> 4;
  const int l16   = lane & 15;
  const int Mbase = blockIdx.x * 128 + wave * 16;
  const int Nbase = blockIdx.y * 64;

  v8f acc[4] = {};
  const unsigned short* aptr = attn + (Mbase + l16) * 256 + half * 8;
  const unsigned short* bptr = Wtf  + (Nbase + l16) * 256 + half * 8;
  __builtin_prefetch(aptr, 0, 1);
  __builtin_prefetch(bptr, 0, 1);

  #pragma unroll
  for (int ks = 0; ks < 8; ++ks) {
    Frag32 a;
    a.q[0] = *(const uint4*)(aptr + ks * 32);
    a.q[1] = *(const uint4*)(aptr + ks * 32 + 16);
    Frag32 b[4];
    #pragma unroll
    for (int j = 0; j < 4; ++j) {
      const unsigned short* bp = bptr + j * 4096 + ks * 32;
      b[j].q[0] = *(const uint4*)(bp);
      b[j].q[1] = *(const uint4*)(bp + 16);
    }
    #pragma unroll
    for (int j = 0; j < 4; ++j)
      acc[j] = __builtin_amdgcn_wmma_f32_16x16x32_bf16(
          false, a.v, false, b[j].v, (short)0, acc[j], false, false);
  }

  #pragma unroll
  for (int j = 0; j < 4; ++j) {
    const int n = Nbase + j * 16 + l16;
    const float bias = bf[n];
    #pragma unroll
    for (int r = 0; r < 8; ++r) {
      const int row = Mbase + r + half * 8;
      out[row * 256 + n] = acc[j][r] + bias;
    }
  }
}

// ---------- launcher ----------
extern "C" void kernel_launch(void* const* d_in, const int* in_sizes, int n_in,
                              void* d_out, int out_size, void* d_ws, size_t ws_size,
                              hipStream_t stream) {
  const float* x1d  = (const float*)d_in[0];
  const float* mask = (const float*)d_in[1];
  const float* ln_g = (const float*)d_in[2];
  const float* ln_b = (const float*)d_in[3];
  const float* Wq   = (const float*)d_in[4];
  const float* Wk   = (const float*)d_in[5];
  const float* Wv   = (const float*)d_in[6];
  const float* Wg   = (const float*)d_in[7];
  const float* bg   = (const float*)d_in[8];
  const float* Wf   = (const float*)d_in[9];
  const float* bf   = (const float*)d_in[10];
  float* out = (float*)d_out;

  char* ws = (char*)d_ws;
  unsigned short* xn   = (unsigned short*)ws;                                  // 32 MiB
  unsigned short* qkvg = (unsigned short*)(ws + (size_t)32  * 1024 * 1024);    // 128 MiB
  unsigned short* Wt   = (unsigned short*)(ws + (size_t)160 * 1024 * 1024);    // 512 KiB
  unsigned short* Wtf  = Wt + 1024 * 256;                                      // 128 KiB
  unsigned short* attn = xn;  // xn is dead after gemm_qkvg; reuse its 32 MiB

  prep_weights<<<1280, 256, 0, stream>>>(Wq, Wk, Wv, Wg, Wf, Wt, Wtf);
  ln_kernel<<<8192, 256, 0, stream>>>(x1d, ln_g, ln_b, xn);
  gemm_qkvg<<<dim3(512, 16), 256, 0, stream>>>(xn, Wt, bg, qkvg);
  attn_kernel<<<dim3(256, 8, 4), 128, 0, stream>>>(qkvg, mask, attn);
  gemm_out<<<dim3(512, 4), 256, 0, stream>>>(attn, Wtf, bf, out);
}
